// SphereTracingRenderer_25151328485581
// MI455X (gfx1250) — matrix operand order repaired and checked by reading, not compiled, not run on hardware
//
#include <hip/hip_runtime.h>
#include <hip/hip_bf16.h>
#include <math.h>

typedef __attribute__((ext_vector_type(16))) _Float16 v16h;
typedef __attribute__((ext_vector_type(8)))  _Float16 v8h;
typedef __attribute__((ext_vector_type(8)))  float    v8f;

#define HIDDEN  256
#define THREADS 128            // 4 wave32 waves per workgroup
#define WAVES   4

// D = A(16x32 f16) * B(32x16 f16) + C(16x16 f32), wave32, EXEC all-1s.
static __device__ __forceinline__ v8f wmma_f16(v16h a, v16h b, v8f c) {
    return __builtin_amdgcn_wmma_f32_16x16x32_f16(false, a, false, b, (short)0, c, false, false);
}

// Fire-and-forget 16B global->LDS copy (ASYNCcnt-tracked), GV addressing mode.
// lds_off is the wave-relative LDS byte offset (= low 32 bits of a __shared__
// generic pointer per the gfx1250 flat-aperture mapping).
static __device__ __forceinline__ void async_copy16(unsigned int lds_off,
                                                    const void* gaddr) {
    asm volatile("global_load_async_to_lds_b128 %0, %1, off"
                 :: "v"(lds_off), "v"((unsigned long long)gaddr)
                 : "memory");
}

static __device__ __forceinline__ void wait_async0() {
    asm volatile("s_wait_asynccnt 0x0" ::: "memory");
}

__global__ __launch_bounds__(THREADS)
void sphere_trace_mlp_kernel(const float* __restrict__ origins,
                             const float* __restrict__ directions,
                             const float* __restrict__ center,
                             const float* __restrict__ radius,
                             const float* __restrict__ W1,
                             const float* __restrict__ b1,
                             const float* __restrict__ W2,
                             const float* __restrict__ b2,
                             const int*   __restrict__ max_iters,
                             float* __restrict__ out)
{
    // f16 weight tiles pre-padded so WMMA K/N overreach multiplies exact zeros.
    __shared__ _Float16 W1p[32 * HIDDEN];        // [K pad 3->32][256]   16 KB
    __shared__ _Float16 W2p[HIDDEN * 16];        // [K=256][N pad 3->16]  8 KB
    __shared__ float    b1s[HIDDEN];             //                       1 KB
    __shared__ float    b2s[4];
    __shared__ float4   Pbuf[THREADS];           // traced point + hit mask
    __shared__ __align__(16) _Float16 Hbuf[WAVES][16 * HIDDEN]; // 32 KB
    // Raw f32 weights are async-DMA'd into Hbuf[0] (7168 B < 8 KB), converted
    // to the padded f16 tiles, then Hbuf is reused as the H activation tile.

    const int tid = threadIdx.x;

    // ---- kick off async global->LDS weight staging (overlaps the trace) ----
    {
        const unsigned int raw0 = (unsigned int)(unsigned long long)&Hbuf[0][0];
        // W1: 3 KB, W2: 3 KB at +3072, b1: 1 KB at +6144 -> 448 x 16B chunks
        for (int c = tid; c < 192; c += THREADS)
            async_copy16(raw0 + c * 16, (const char*)W1 + c * 16);
        for (int c = tid; c < 192; c += THREADS)
            async_copy16(raw0 + 3072 + c * 16, (const char*)W2 + c * 16);
        if (tid < 64)
            async_copy16(raw0 + 6144 + tid * 16, (const char*)b1 + tid * 16);
    }
    if (tid < 4) b2s[tid] = (tid < 3) ? b2[tid] : 0.f;   // 12 B: plain load

    // ---- sphere trace: one ray per lane (serial sqrt chain -> keep lanes hot,
    //      async weight DMA lands underneath this) ----
    const int g = blockIdx.x * THREADS + tid;
    const float ox = origins[3*g+0], oy = origins[3*g+1], oz = origins[3*g+2];
    const float dx = directions[3*g+0], dy = directions[3*g+1], dz = directions[3*g+2];
    const float cx = center[0], cy = center[1], cz = center[2];
    const float r  = radius[0];
    const int iters = max_iters[0] - 1;          // torch loop: 63 effective t-updates
    float t = 0.f;
    for (int it = 0; it < iters; ++it) {
        float px = ox + t*dx - cx;
        float py = oy + t*dy - cy;
        float pz = oz + t*dz - cz;
        t += __builtin_amdgcn_sqrtf(px*px + py*py + pz*pz) - r;
    }
    const float px = ox + t*dx, py = oy + t*dy, pz = oz + t*dz;
    const float qx = px-cx, qy = py-cy, qz = pz-cz;
    const float sd = __builtin_amdgcn_sqrtf(qx*qx + qy*qy + qz*qz) - r;
    Pbuf[tid] = make_float4(px, py, pz, (sd < 1e-8f) ? 1.f : 0.f);

    // ---- async DMA done? then convert raw f32 -> padded f16 tiles ----
    wait_async0();
    __syncthreads();
    {
        const float* rawW1 = (const float*)&Hbuf[0][0];
        const float* rawW2 = rawW1 + 768;
        const float* rawb1 = rawW1 + 1536;
        for (int idx = tid; idx < 32 * HIDDEN; idx += THREADS) {
            int k = idx >> 8, n = idx & 255;
            W1p[idx] = (k < 3) ? (_Float16)rawW1[k * HIDDEN + n] : (_Float16)0.f;
        }
        for (int idx = tid; idx < HIDDEN * 16; idx += THREADS) {
            int k = idx >> 4, n = idx & 15;
            W2p[idx] = (n < 3) ? (_Float16)rawW2[k * 3 + n] : (_Float16)0.f;
        }
        for (int idx = tid; idx < HIDDEN; idx += THREADS) b1s[idx] = rawb1[idx];
    }
    __syncthreads();

    // ---- MLP: two 16-ray WMMA tiles per wave ----
    const int  w  = tid >> 5;
    const int  L  = tid & 31;
    const int  m  = L & 15;          // matrix row (A) / column N (C)
    const bool hi = (L >= 16);
    _Float16* Hrow = &Hbuf[w][0];

    for (int s = 0; s < 2; ++s) {
        // A1 = point tile 16x32 f16: K=0,1,2 -> (x,y,z), rest zero.
        // A layout: lane L row m=L&15; elem i -> K=(i<8?i:i+8)+(hi?8:0),
        // so only lo-lanes, elems 0..2, carry data.
        v16h a1 = {};
        if (!hi) {
            float4 p = Pbuf[w*32 + s*16 + m];
            a1[0] = (_Float16)p.x; a1[1] = (_Float16)p.y; a1[2] = (_Float16)p.z;
        }

        // GEMM1: H[16,256] = relu(A1 * W1 + b1), 16 N-chunks of 16
        #pragma unroll 4
        for (int j = 0; j < 16; ++j) {
            // B operand: lane L = column N=m, elem i -> K=(i<8?i:i+8)+(hi?8:0)
            const _Float16* bp = &W1p[j * 16 + m];
            v16h bmat;
            #pragma unroll
            for (int i = 0; i < 16; ++i)
                bmat[i] = bp[((i < 8 ? i : i + 8) + (hi ? 8 : 0)) * HIDDEN];
            v8f c = {};
            c = wmma_f16(a1, bmat, c);
            // C layout: VGPR v -> M=v+(hi?8:0), N=m; bias is per-N
            const float bias = b1s[j * 16 + m];
            #pragma unroll
            for (int v = 0; v < 8; ++v) {
                float h = c[v] + bias;
                h = h > 0.f ? h : 0.f;
                const int mv = v + (hi ? 8 : 0);
                Hrow[mv * HIDDEN + j * 16 + m] = (_Float16)h;
            }
        }
        __syncthreads();   // H tile visible across lanes before A2 gathers

        // GEMM2: OUT[16,16] = H[16,256] * W2pad[256,16], 8 accumulating WMMAs
        v8f acc = {};
        #pragma unroll
        for (int c8 = 0; c8 < 8; ++c8) {
            // A operand from row-major H: two contiguous 16B runs per lane
            const _Float16* ap = Hrow + m * HIDDEN + c8 * 32 + (hi ? 8 : 0);
            v8h alo = *(const v8h*)ap;          // K = 32c + off .. +7
            v8h ahi = *(const v8h*)(ap + 16);   // K = 32c + 16 + off .. +7
            v16h amat = __builtin_shufflevector(alo, ahi,
                0,1,2,3,4,5,6,7,8,9,10,11,12,13,14,15);
            const _Float16* bp = &W2p[(c8 * 32) * 16 + m];
            v16h bmat;
            #pragma unroll
            for (int i = 0; i < 16; ++i)
                bmat[i] = bp[((i < 8 ? i : i + 8) + (hi ? 8 : 0)) * 16];
            acc = wmma_f16(amat, bmat, acc);
        }

        // Epilogue: N<3 lanes hold real color channels; bias + sigmoid + mask.
        if (m < 3) {
            const float bias = b2s[m];
            #pragma unroll
            for (int v = 0; v < 8; ++v) {
                const int   mv   = v + (hi ? 8 : 0);
                const int   ray  = blockIdx.x * THREADS + w * 32 + s * 16 + mv;
                const float x    = acc[v] + bias;
                const float sig  = 1.f / (1.f + __expf(-x));
                const float mask = Pbuf[w * 32 + s * 16 + mv].w;
                out[(size_t)ray * 3 + m] = sig * mask;
            }
        }
        __syncthreads();   // protect Hbuf before next tile overwrites it
    }
}

extern "C" void kernel_launch(void* const* d_in, const int* in_sizes, int n_in,
                              void* d_out, int out_size, void* d_ws, size_t ws_size,
                              hipStream_t stream) {
    const float* origins    = (const float*)d_in[0];
    const float* directions = (const float*)d_in[1];
    const float* center     = (const float*)d_in[2];
    const float* radius     = (const float*)d_in[3];
    const float* W1         = (const float*)d_in[4];
    const float* b1         = (const float*)d_in[5];
    const float* W2         = (const float*)d_in[6];
    const float* b2         = (const float*)d_in[7];
    const int*   max_iters  = (const int*)d_in[8];
    float* out = (float*)d_out;

    const int n_rays = in_sizes[0] / 3;          // 262144
    const int blocks = n_rays / THREADS;         // 2048 (divides exactly)

    sphere_trace_mlp_kernel<<<blocks, THREADS, 0, stream>>>(
        origins, directions, center, radius, W1, b1, W2, b2, max_iters, out);
}
